// NeighborhoodAttention_34608846471407
// MI455X (gfx1250) — matrix operand order repaired and checked by reading, not compiled, not run on hardware
//
#include <hip/hip_runtime.h>

#define N_NODES 50000
#define N_EDGES 800000

typedef __attribute__((ext_vector_type(16))) _Float16 v16h;
typedef __attribute__((ext_vector_type(8)))  _Float16 v8h;
typedef __attribute__((ext_vector_type(8)))  float    v8f;

static __device__ __forceinline__ unsigned f2ord(float f){
  unsigned u = __float_as_uint(f);
  return (u & 0x80000000u) ? ~u : (u | 0x80000000u);
}
static __device__ __forceinline__ float ord2f(unsigned u){
  unsigned v = (u & 0x80000000u) ? (u & 0x7FFFFFFFu) : ~u;
  return __uint_as_float(v);
}

// Build a 16-bit WMMA A-fragment from LDS row `p` (points at row_base + 32*kt + hioff).
// Per ISA A-layout: lane-lo holds K 0-7 & 16-23, lane-hi holds K 8-15 & 24-31.
static __device__ __forceinline__ v16h load_a_frag(const _Float16* p){
  v8h lo = *(const v8h*)p;
  v8h hi = *(const v8h*)(p + 16);
  v16h a;
#pragma unroll
  for (int i = 0; i < 8; ++i){ a[i] = lo[i]; a[i+8] = hi[i]; }
  return a;
}

#define WMMA16(a,b,c) __builtin_amdgcn_wmma_f32_16x16x32_f16(false,(a),false,(b),(short)0,(c),false,false)

// Repack fp32 weight [K,128] row-major -> f16 WMMA B-fragments.
// B layout (32x16 f16): lanes 0-15 hold K 0-7 & 16-23, lanes 16-31 hold K 8-15 & 24-31; col = n*16 + lane%16.
// Fragment element i of lane -> K = kt*32 + (i>=8?16:0) + (lane>=16?8:0) + (i&7).
__global__ void repack_w(const float* __restrict__ W, _Float16* __restrict__ out, int total){
  int t = blockIdx.x * 256 + threadIdx.x;
  if (t >= total) return;
  int i    = t & 15;
  int lane = (t >> 4) & 31;
  int n    = (t >> 9) & 7;
  int kt   = t >> 12;
  int K    = kt*32 + ((i >> 3) << 4) + ((lane >> 4) << 3) + (i & 7);
  int col  = n*16 + (lane & 15);
  out[t] = (_Float16)W[K*128 + col];
}

__global__ void init_buffers(unsigned* __restrict__ segmax, float* __restrict__ segsum,
                             float* __restrict__ aggr){
  int i = blockIdx.x * 256 + threadIdx.x;
  if (i < N_NODES*8){ segmax[i] = 0u; segsum[i] = 0.f; }
  if (i < N_NODES*128) aggr[i] = 0.f;
}

// One wave = 16 edges x 128 outputs. Computes k (-> scores + atomic seg-max) and v (-> f16 ws).
__global__ __launch_bounds__(128) void edge_kernel(
    const float* __restrict__ xsrc, const float* __restrict__ xdst,
    const float* __restrict__ eattr,
    const long long* __restrict__ srci, const long long* __restrict__ dsti,
    const float* __restrict__ q,
    const _Float16* __restrict__ kW0f, const float* __restrict__ kb0,
    const _Float16* __restrict__ kW1f, const float* __restrict__ kb1,
    const _Float16* __restrict__ vW0f, const float* __restrict__ vb0,
    const _Float16* __restrict__ vW1f, const float* __restrict__ vb1,
    float* __restrict__ scores, unsigned* __restrict__ segmax,
    _Float16* __restrict__ vws)
{
  __shared__ __attribute__((aligned(16))) _Float16 sX[4][16][160];
  __shared__ __attribute__((aligned(16))) _Float16 sH[4][16][128];
  const int lane  = threadIdx.x & 31;
  const int w     = threadIdx.x >> 5;
  const long long e0 = ((long long)blockIdx.x * 4 + w) * 16;
  const int col   = lane & 15;
  const int hioff = (lane >> 4) * 8;

  // Gather [x_src | x_dst | edge_attr] -> f16 LDS tile [16][160]
  for (int e = 0; e < 16; ++e){
    long long ge = e0 + e;
    long long si = srci[ge], di = dsti[ge];
    for (int f = lane; f < 160; f += 32){
      float v;
      if (f < 64)       v = xsrc[si*64 + f];
      else if (f < 128) v = xdst[di*64 + (f - 64)];
      else              v = eattr[ge*32 + (f - 128)];
      sX[w][e][f] = (_Float16)v;
    }
  }
  __syncthreads();

  v8f acc[8], acc2[8];

  // ======================= K path: h = relu(x@W0+b0); k = h + h@W1 + b1 =======================
#pragma unroll
  for (int n = 0; n < 8; ++n){
    float b = kb0[n*16 + col];
#pragma unroll
    for (int r = 0; r < 8; ++r) acc[n][r] = b;
  }
#pragma unroll
  for (int kt = 0; kt < 5; ++kt){
    v16h a = load_a_frag(&sX[w][col][kt*32 + hioff]);
#pragma unroll
    for (int n = 0; n < 8; ++n){
      v16h b = *(const v16h*)(kW0f + (((kt*8 + n)*32 + lane) << 4));
      acc[n] = WMMA16(a, b, acc[n]);
    }
  }
  __syncthreads();
#pragma unroll
  for (int n = 0; n < 8; ++n){
    float b1 = kb1[n*16 + col];
#pragma unroll
    for (int r = 0; r < 8; ++r){
      float h = acc[n][r]; h = h > 0.f ? h : 0.f;
      sH[w][r + hioff][n*16 + col] = (_Float16)h;   // C layout: row = r + hi*8, col = n*16+col
      acc2[n][r] = h + b1;                          // residual + bias pre-loaded into accumulator
    }
  }
  __syncthreads();
#pragma unroll
  for (int kt = 0; kt < 4; ++kt){
    v16h a = load_a_frag(&sH[w][col][kt*32 + hioff]);
#pragma unroll
    for (int n = 0; n < 8; ++n){
      v16h b = *(const v16h*)(kW1f + (((kt*8 + n)*32 + lane) << 4));
      acc2[n] = WMMA16(a, b, acc2[n]);
    }
  }

  // scores: head n == N-tile n (DH = 16). Reduce q.k over the 16 columns via shfl.
#pragma unroll
  for (int n = 0; n < 8; ++n){
    float qv = q[n*16 + col];
#pragma unroll
    for (int r = 0; r < 8; ++r){
      float p = qv * acc2[n][r];
      p += __shfl_xor(p, 1, 32);
      p += __shfl_xor(p, 2, 32);
      p += __shfl_xor(p, 4, 32);
      p += __shfl_xor(p, 8, 32);
      if (col == 0){
        long long ge = e0 + r + hioff;
        float s = p * 0.25f;                        // 1/sqrt(16)
        scores[ge*8 + n] = s;
        atomicMax(&segmax[dsti[ge]*8 + n], f2ord(s));
      }
    }
  }

  // ======================= V path (same stack, vW*) =======================
#pragma unroll
  for (int n = 0; n < 8; ++n){
    float b = vb0[n*16 + col];
#pragma unroll
    for (int r = 0; r < 8; ++r) acc[n][r] = b;
  }
#pragma unroll
  for (int kt = 0; kt < 5; ++kt){
    v16h a = load_a_frag(&sX[w][col][kt*32 + hioff]);
#pragma unroll
    for (int n = 0; n < 8; ++n){
      v16h b = *(const v16h*)(vW0f + (((kt*8 + n)*32 + lane) << 4));
      acc[n] = WMMA16(a, b, acc[n]);
    }
  }
  __syncthreads();
#pragma unroll
  for (int n = 0; n < 8; ++n){
    float b1 = vb1[n*16 + col];
#pragma unroll
    for (int r = 0; r < 8; ++r){
      float h = acc[n][r]; h = h > 0.f ? h : 0.f;
      sH[w][r + hioff][n*16 + col] = (_Float16)h;
      acc2[n][r] = h + b1;
    }
  }
  __syncthreads();
#pragma unroll
  for (int kt = 0; kt < 4; ++kt){
    v16h a = load_a_frag(&sH[w][col][kt*32 + hioff]);
#pragma unroll
    for (int n = 0; n < 8; ++n){
      v16h b = *(const v16h*)(vW1f + (((kt*8 + n)*32 + lane) << 4));
      acc2[n] = WMMA16(a, b, acc2[n]);
    }
  }
#pragma unroll
  for (int n = 0; n < 8; ++n){
#pragma unroll
    for (int r = 0; r < 8; ++r){
      long long ge = e0 + r + hioff;
      vws[ge*128 + n*16 + col] = (_Float16)acc2[n][r];
    }
  }
}

__global__ void exp_sum_kernel(const long long* __restrict__ dsti,
                               const unsigned* __restrict__ segmax,
                               float* __restrict__ sc,       // in: scores, out: exp (in place)
                               float* __restrict__ segsum)
{
  long long i = (long long)blockIdx.x * 256 + threadIdx.x;
  if (i >= (long long)N_EDGES * 8) return;
  long long e = i >> 3; int h = (int)(i & 7);
  long long d = dsti[e];
  float m  = ord2f(segmax[d*8 + h]);
  float ex = __expf(sc[i] - m);
  sc[i] = ex;
  atomicAdd(&segsum[d*8 + h], ex);
}

__global__ void message_kernel(const long long* __restrict__ dsti,
                               const float* __restrict__ ex,
                               const float* __restrict__ segsum,
                               const _Float16* __restrict__ vws,
                               float* __restrict__ aggr)
{
  long long i = (long long)blockIdx.x * 256 + threadIdx.x;
  if (i >= (long long)N_EDGES * 128) return;
  long long e = i >> 7; int dcol = (int)(i & 127); int h = dcol >> 4;
  long long d = dsti[e];
  float alpha = ex[e*8 + h] / segsum[d*8 + h];
  atomicAdd(&aggr[d*128 + dcol], alpha * (float)vws[i]);
}

// out = relu( h + h@oW1 + ob1 ), h = relu( relu(aggr)@oW0 + ob0 ); 16 nodes per wave.
__global__ __launch_bounds__(128) void out_kernel(
    const float* __restrict__ aggr,
    const _Float16* __restrict__ oW0f, const float* __restrict__ ob0,
    const _Float16* __restrict__ oW1f, const float* __restrict__ ob1,
    float* __restrict__ out)
{
  __shared__ __attribute__((aligned(16))) _Float16 sX[4][16][128];
  __shared__ __attribute__((aligned(16))) _Float16 sH[4][16][128];
  const int lane  = threadIdx.x & 31;
  const int w     = threadIdx.x >> 5;
  const int tile  = blockIdx.x * 4 + w;
  const bool act  = tile < (N_NODES / 16);
  const long long n0 = (long long)tile * 16;
  const int col   = lane & 15;
  const int hioff = (lane >> 4) * 8;

  for (int e = 0; e < 16; ++e)
    for (int f = lane; f < 128; f += 32){
      float v = act ? aggr[(n0 + e)*128 + f] : 0.f;
      v = v > 0.f ? v : 0.f;                         // relu before out-stack
      sX[w][e][f] = (_Float16)v;
    }
  __syncthreads();

  v8f acc[8], acc2[8];
#pragma unroll
  for (int n = 0; n < 8; ++n){
    float b = ob0[n*16 + col];
#pragma unroll
    for (int r = 0; r < 8; ++r) acc[n][r] = b;
  }
#pragma unroll
  for (int kt = 0; kt < 4; ++kt){
    v16h a = load_a_frag(&sX[w][col][kt*32 + hioff]);
#pragma unroll
    for (int n = 0; n < 8; ++n){
      v16h b = *(const v16h*)(oW0f + (((kt*8 + n)*32 + lane) << 4));
      acc[n] = WMMA16(a, b, acc[n]);
    }
  }
  __syncthreads();
#pragma unroll
  for (int n = 0; n < 8; ++n){
    float b1 = ob1[n*16 + col];
#pragma unroll
    for (int r = 0; r < 8; ++r){
      float h = acc[n][r]; h = h > 0.f ? h : 0.f;
      sH[w][r + hioff][n*16 + col] = (_Float16)h;
      acc2[n][r] = h + b1;
    }
  }
  __syncthreads();
#pragma unroll
  for (int kt = 0; kt < 4; ++kt){
    v16h a = load_a_frag(&sH[w][col][kt*32 + hioff]);
#pragma unroll
    for (int n = 0; n < 8; ++n){
      v16h b = *(const v16h*)(oW1f + (((kt*8 + n)*32 + lane) << 4));
      acc2[n] = WMMA16(a, b, acc2[n]);
    }
  }
  if (act){
#pragma unroll
    for (int n = 0; n < 8; ++n)
#pragma unroll
      for (int r = 0; r < 8; ++r){
        float y = acc2[n][r]; y = y > 0.f ? y : 0.f;  // last_activation=True
        out[(n0 + r + hioff)*128 + n*16 + col] = y;
      }
  }
}

extern "C" void kernel_launch(void* const* d_in, const int* in_sizes, int n_in,
                              void* d_out, int out_size, void* d_ws, size_t ws_size,
                              hipStream_t stream)
{
  (void)in_sizes; (void)n_in; (void)out_size; (void)ws_size;
  const float* xsrc  = (const float*)d_in[0];
  const float* xdst  = (const float*)d_in[1];
  const float* eattr = (const float*)d_in[2];
  const long long* eidx = (const long long*)d_in[3];
  const float* q   = (const float*)d_in[4];
  const float* kW0 = (const float*)d_in[5];
  const float* kb0 = (const float*)d_in[6];
  const float* kW1 = (const float*)d_in[7];
  const float* kb1 = (const float*)d_in[8];
  const float* vW0 = (const float*)d_in[9];
  const float* vb0 = (const float*)d_in[10];
  const float* vW1 = (const float*)d_in[11];
  const float* vb1 = (const float*)d_in[12];
  const float* oW0 = (const float*)d_in[13];
  const float* ob0 = (const float*)d_in[14];
  const float* oW1 = (const float*)d_in[15];
  const float* ob1 = (const float*)d_in[16];

  const long long* srci = eidx;
  const long long* dsti = eidx + N_EDGES;

  char* ws = (char*)d_ws;
  size_t o = 0;
  auto alloc = [&](size_t bytes){ char* p = ws + o; o += (bytes + 255) & ~(size_t)255; return p; };
  _Float16* kW0f = (_Float16*)alloc(160*128*2);
  _Float16* kW1f = (_Float16*)alloc(128*128*2);
  _Float16* vW0f = (_Float16*)alloc(160*128*2);
  _Float16* vW1f = (_Float16*)alloc(128*128*2);
  _Float16* oW0f = (_Float16*)alloc(128*128*2);
  _Float16* oW1f = (_Float16*)alloc(128*128*2);
  unsigned* segmax = (unsigned*)alloc((size_t)N_NODES*8*4);
  float*    segsum = (float*)   alloc((size_t)N_NODES*8*4);
  float*    aggr   = (float*)   alloc((size_t)N_NODES*128*4);
  float*    sc     = (float*)   alloc((size_t)N_EDGES*8*4);     // scores, then exp (in place)
  _Float16* vws    = (_Float16*)alloc((size_t)N_EDGES*128*2);

  repack_w<<<(160*128 + 255)/256, 256, 0, stream>>>(kW0, kW0f, 160*128);
  repack_w<<<(128*128 + 255)/256, 256, 0, stream>>>(kW1, kW1f, 128*128);
  repack_w<<<(160*128 + 255)/256, 256, 0, stream>>>(vW0, vW0f, 160*128);
  repack_w<<<(128*128 + 255)/256, 256, 0, stream>>>(vW1, vW1f, 128*128);
  repack_w<<<(128*128 + 255)/256, 256, 0, stream>>>(oW0, oW0f, 128*128);
  repack_w<<<(128*128 + 255)/256, 256, 0, stream>>>(oW1, oW1f, 128*128);

  init_buffers<<<(N_NODES*128 + 255)/256, 256, 0, stream>>>(segmax, segsum, aggr);

  edge_kernel<<<N_EDGES/(16*4), 128, 0, stream>>>(
      xsrc, xdst, eattr, srci, dsti, q,
      kW0f, kb0, kW1f, kb1, vW0f, vb0, vW1f, vb1,
      sc, segmax, vws);

  exp_sum_kernel<<<(int)(((long long)N_EDGES*8 + 255)/256), 256, 0, stream>>>(dsti, segmax, sc, segsum);

  message_kernel<<<(int)(((long long)N_EDGES*128 + 255)/256), 256, 0, stream>>>(dsti, sc, segsum, vws, aggr);

  out_kernel<<<(N_NODES/16 + 3)/4, 128, 0, stream>>>(aggr, oW0f, ob0, oW1f, ob1, (float*)d_out);
}
